// GATParser_20426864459822
// MI455X (gfx1250) — matrix operand order repaired and checked by reading, not compiled, not run on hardware
//
#include <hip/hip_runtime.h>
#include <hip/hip_bf16.h>
#include <math.h>

// ---------------------------------------------------------------------------
// GATv2 biaffine parser forward for MI455X (gfx1250, wave32).
//  * All GEMMs: v_wmma_f32_16x16x32_bf16, f32 accumulate.
//  * Operands pre-converted to bf16 (weights transposed to [N,K] mirrors,
//    buffers padded) so GEMM tile staging is pure byte movement:
//    global_load_async_to_lds_b128 (ASYNCcnt), double-buffered LDS,
//    1 barrier per k-step, NO edge fallback path.
//  * Block tile 64x64, wave tile 16x32: one A fragment feeds 2 WMMAs.
//  * Fragments contiguous in LDS: 2x ds_load_b128 per fragment.
//  * Graph ops (softmax / top-k / GAT scatter): wave32 VALU + float atomics.
// ---------------------------------------------------------------------------

typedef __bf16 bfloat16_t;
typedef __attribute__((ext_vector_type(8)))  __bf16 v8bf;
typedef __attribute__((ext_vector_type(16))) __bf16 v16bf;
typedef __attribute__((ext_vector_type(8)))  float  v8f;

// problem constants
#define CB   32
#define CL   256
#define CD   1024
#define CA   512
#define CT   128
#define CH   4
#define CK   3
#define CS   257
#define CE   771          // S*K
#define CFP  544          // padded (A+1)=513 -> next mult of 32 (K-padded)
#define LBS  (576LL*544)  // bilinear-W^T slice: 513->576 rows, 544 K-stride
#define ROWS (32LL * 257LL)
#define PADR 64           // M/N over-read padding rows for tile reads

// ---------------- wave reductions (wave32) ----------------
__device__ inline float wave_max_f(float v) {
    for (int o = 16; o; o >>= 1) v = fmaxf(v, __shfl_xor(v, o, 32));
    return v;
}
__device__ inline float wave_sum_f(float v) {
    for (int o = 16; o; o >>= 1) v += __shfl_xor(v, o, 32);
    return v;
}

// ordered-uint encoding so atomicMax(uint) == float max
__device__ inline unsigned enc_f(float f) {
    unsigned u = __float_as_uint(f);
    return (u & 0x80000000u) ? ~u : (u | 0x80000000u);
}
__device__ inline float dec_f(unsigned e) {
    unsigned u = (e & 0x80000000u) ? (e & 0x7FFFFFFFu) : ~e;
    return __uint_as_float(u);
}

// ---------------------------------------------------------------------------
// GEMM: C[b] = act(A[b] @ BT[b]^T + bias)
//  A:  bf16 [M,K] row-major, lda mult of 8; rows over-readable to tile edge
//  BT: bf16 [N,K] row-major, ldb mult of 8; rows over-readable; K-pad zeroed
//  C:  f32 out (optional), Cb: bf16 mirror (optional); stores guarded.
// Block 256 thr = 8 waves -> 64x64 tile; wave 16x32 (2 wmma); KT=32;
// double-buffered LDS fed exclusively by global_load_async_to_lds_b128.
// biasMode: 0 none, 1 per-col (f32), 2 scalar bias[0]. act: 0 none, 1 ELU.
// ---------------------------------------------------------------------------
#define TM 64
#define TN 64
#define KT 32
#define LP 40   // LDS row pitch in bf16 (80B, 16B aligned)

__global__ __launch_bounds__(256)
void gemm_wmma_bf16(const bfloat16_t* __restrict__ A,
                    const bfloat16_t* __restrict__ BT,
                    const float* __restrict__ bias,
                    float* __restrict__ C, bfloat16_t* __restrict__ Cb,
                    int M, int N, int Kd, int lda, int ldb, int ldc,
                    long long sA, long long sB, long long sC,
                    int biasMode, int act)
{
    __shared__ bfloat16_t As[2][TM][LP];
    __shared__ bfloat16_t Bs[2][TN][LP];

    const int bz = blockIdx.z;
    A  += (long long)bz * sA;
    BT += (long long)bz * sB;

    const int tid  = threadIdx.x;
    const int lane = tid & 31;
    const int wave = tid >> 5;
    const int wm   = wave >> 1;          // 0..3  (16-row slab)
    const int wn   = wave & 1;           // 0..1  (32-col slab)
    const int bm   = blockIdx.y * TM;
    const int bn   = blockIdx.x * TN;
    const int hlf  = lane >> 4;
    const int lr   = lane & 15;

    // each thread stages one 16B chunk of A and one of B (64 rows x 4 chunks)
    const int r16 = tid >> 2, c8 = (tid & 3) * 8;

    auto stage = [&](int buf, int k0) {
        unsigned la = (unsigned)(size_t)&As[buf][r16][c8];
        unsigned long long ga =
            (unsigned long long)(size_t)(A + (long long)(bm + r16) * lda + k0 + c8);
        asm volatile("global_load_async_to_lds_b128 %0, %1, off"
                     :: "v"(la), "v"(ga) : "memory");
        unsigned lb = (unsigned)(size_t)&Bs[buf][r16][c8];
        unsigned long long gb =
            (unsigned long long)(size_t)(BT + (long long)(bn + r16) * ldb + k0 + c8);
        asm volatile("global_load_async_to_lds_b128 %0, %1, off"
                     :: "v"(lb), "v"(gb) : "memory");
    };

    v8f acc0 = {}, acc1 = {};
    const int nk = (Kd + KT - 1) / KT;

    stage(0, 0);

    for (int t = 0; t < nk; ++t) {
        asm volatile("s_wait_asynccnt 0" ::: "memory");
        __syncthreads();
        const int cur = t & 1;
        if (t + 1 < nk)                   // prefetch next tile while computing
            stage(cur ^ 1, (t + 1) * KT);

        // A fragment: row wm*16+lr, K run [hlf*16,+16)
        const v8bf* arow = (const v8bf*)(&As[cur][wm * 16 + lr][0]);
        v16bf af = __builtin_shufflevector(arow[2 * hlf], arow[2 * hlf + 1],
                                           0, 1, 2, 3, 4, 5, 6, 7,
                                           8, 9, 10, 11, 12, 13, 14, 15);
        // B fragments: rows (N) wn*32+lr and wn*32+16+lr, K run [hlf*16,+16)
        const v8bf* b0row = (const v8bf*)(&Bs[cur][wn * 32 + lr][0]);
        v16bf bf0 = __builtin_shufflevector(b0row[2 * hlf], b0row[2 * hlf + 1],
                                            0, 1, 2, 3, 4, 5, 6, 7,
                                            8, 9, 10, 11, 12, 13, 14, 15);
        const v8bf* b1row = (const v8bf*)(&Bs[cur][wn * 32 + 16 + lr][0]);
        v16bf bf1 = __builtin_shufflevector(b1row[2 * hlf], b1row[2 * hlf + 1],
                                            0, 1, 2, 3, 4, 5, 6, 7,
                                            8, 9, 10, 11, 12, 13, 14, 15);

        acc0 = __builtin_amdgcn_wmma_f32_16x16x32_bf16(
                   false, af, false, bf0, (short)0, acc0, false, false);
        acc1 = __builtin_amdgcn_wmma_f32_16x16x32_bf16(
                   false, af, false, bf1, (short)0, acc1, false, false);
    }

    // epilogue: VGPR r holds row (hlf*8 + r), col = lane&15
    const int gn0 = bn + wn * 32 + lr;
    const int gn1 = gn0 + 16;
#pragma unroll
    for (int r = 0; r < 8; ++r) {
        int gm = bm + wm * 16 + hlf * 8 + r;
        if (gm < M) {
            long long ro = (long long)bz * sC + (long long)gm * ldc;
            if (gn0 < N) {
                float v = acc0[r];
                if (biasMode == 1)      v += bias[gn0];
                else if (biasMode == 2) v += bias[0];
                if (act == 1) v = (v > 0.0f) ? v : (expf(v) - 1.0f);
                if (C)  C[ro + gn0]  = v;
                if (Cb) Cb[ro + gn0] = (bfloat16_t)v;
            }
            if (gn1 < N) {
                float v = acc1[r];
                if (biasMode == 1)      v += bias[gn1];
                else if (biasMode == 2) v += bias[0];
                if (act == 1) v = (v > 0.0f) ? v : (expf(v) - 1.0f);
                if (C)  C[ro + gn1]  = v;
                if (Cb) Cb[ro + gn1] = (bfloat16_t)v;
            }
        }
    }
}

// ---------------- conversion / utility kernels ----------------
__global__ void fill_f32(float* p, float v, long long n) {
    long long i = (long long)blockIdx.x * 256 + threadIdx.x;
    if (i < n) p[i] = v;
}
__global__ void fill_u32(unsigned* p, unsigned v, long long n) {
    long long i = (long long)blockIdx.x * 256 + threadIdx.x;
    if (i < n) p[i] = v;
}
__global__ void fill_bf16(bfloat16_t* p, float v, long long n) {
    long long i = (long long)blockIdx.x * 256 + threadIdx.x;
    if (i < n) p[i] = (bfloat16_t)v;
}
__global__ void copy_f32(float* dst, const float* src, long long n) {
    long long i = (long long)blockIdx.x * 256 + threadIdx.x;
    if (i < n) dst[i] = src[i];
}
// dst[n*ldbt + k] = (bf16) src[k*N + n]  (weight [K,N] -> transposed mirror)
__global__ void conv_transpose_bf16(const float* __restrict__ src,
                                    bfloat16_t* __restrict__ dst,
                                    int K, int N, int ldbt) {
    long long i = (long long)blockIdx.x * 256 + threadIdx.x;
    long long tot = (long long)N * ldbt;
    if (i >= tot) return;
    int k = (int)(i % ldbt);
    long long n = i / ldbt;
    dst[i] = (k < K) ? (bfloat16_t)src[(long long)k * N + n] : (bfloat16_t)0.0f;
}

// prepend sentinel row, emit bf16
__global__ void build_xa(const float* __restrict__ x, const float* __restrict__ sent,
                         bfloat16_t* __restrict__ xab) {
    long long i = (long long)blockIdx.x * 256 + threadIdx.x;
    long long tot = ROWS * CD;
    if (i >= tot) return;
    int d = (int)(i % CD);
    long long t = i / CD;
    int s = (int)(t % CS);
    long long b = t / CS;
    float f = (s == 0) ? sent[d] : x[(b * CL + (s - 1)) * CD + d];
    xab[i] = (bfloat16_t)f;
}

__global__ void build_aux(const int* __restrict__ mask, const int* __restrict__ hin,
                          float* __restrict__ mf, int* __restrict__ hi) {
    long long i = (long long)blockIdx.x * 256 + threadIdx.x;
    if (i >= ROWS) return;
    int s = (int)(i % CS);
    long long b = i / CS;
    mf[i] = (s == 0) ? 1.0f : (float)mask[b * CL + (s - 1)];
    hi[i] = (s == 0) ? 0 : hin[b * CL + (s - 1)];
}

// [rows,F] f32 -> [rows,FP] bf16 with col F = 1.0, cols > F = 0 (K-pad zeros)
__global__ void pad_ones(const float* __restrict__ src, bfloat16_t* __restrict__ dst,
                         int F, int FP, long long rows) {
    long long i = (long long)blockIdx.x * 256 + threadIdx.x;
    long long tot = rows * FP;
    if (i >= tot) return;
    int c = (int)(i % FP);
    long long r = i / FP;
    float f = (c < F) ? src[r * F + c] : (c == F ? 1.0f : 0.0f);
    dst[i] = (bfloat16_t)f;
}

// one wave per row: probs = softmax(att) over last axis (S)
__global__ __launch_bounds__(256)
void row_softmax(const float* __restrict__ att, float* __restrict__ probs) {
    long long row = (long long)blockIdx.x * 8 + (threadIdx.x >> 5);
    if (row >= ROWS) return;
    int lane = threadIdx.x & 31;
    const float* r = att + row * CS;
    float* p = probs + row * CS;
    float mx = -3.4e38f;
    for (int j = lane; j < CS; j += 32) mx = fmaxf(mx, r[j]);
    mx = wave_max_f(mx);
    float s = 0.0f;
    for (int j = lane; j < CS; j += 32) s += expf(r[j] - mx);
    s = wave_sum_f(s);
    float inv = 1.0f / s;
    for (int j = lane; j < CS; j += 32) p[j] = expf(r[j] - mx) * inv;
}

// one wave per row: masked log-softmax gathered at hi; accumulate -sum
__global__ __launch_bounds__(256)
void loss_rows(const float* __restrict__ att, const float* __restrict__ mf,
               const int* __restrict__ hi, float* __restrict__ accum) {
    long long row = (long long)blockIdx.x * 8 + (threadIdx.x >> 5);
    if (row >= ROWS) return;
    int lane = threadIdx.x & 31;
    long long b = row / CS;
    int i = (int)(row % CS);
    const float* r  = att + row * CS;
    const float* mb = mf + b * CS;
    float mx = -3.4e38f;
    for (int j = lane; j < CS; j += 32) mx = fmaxf(mx, r[j] + logf(mb[j] + 1e-45f));
    mx = wave_max_f(mx);
    float s = 0.0f;
    for (int j = lane; j < CS; j += 32) s += expf(r[j] + logf(mb[j] + 1e-45f) - mx);
    s = wave_sum_f(s);
    if (lane == 0 && i >= 1) {
        int js = hi[row];
        float z   = r[js] + logf(mb[js] + 1e-45f);
        float val = (z - (mx + logf(s))) * mb[js];
        atomicAdd(accum, -val);
    }
}

// top-3 per row of probs (descending, earliest index wins ties)
__global__ void topk_rows(const float* __restrict__ probs, float* __restrict__ ea,
                          int* __restrict__ ei) {
    long long t = (long long)blockIdx.x * 256 + threadIdx.x;
    if (t >= ROWS) return;
    const float* pr = probs + t * CS;
    float v0 = -1.f, v1 = -1.f, v2 = -1.f;
    int   i0 = 0,    i1 = 0,    i2 = 0;
    for (int j = 0; j < CS; ++j) {
        float f = pr[j];
        if (f > v0)      { v2 = v1; i2 = i1; v1 = v0; i1 = i0; v0 = f; i0 = j; }
        else if (f > v1) { v2 = v1; i2 = i1; v1 = f;  i1 = j; }
        else if (f > v2) { v2 = f;  i2 = j; }
    }
    ea[t * 3] = v0; ea[t * 3 + 1] = v1; ea[t * 3 + 2] = v2;
    ei[t * 3] = i0; ei[t * 3 + 1] = i1; ei[t * 3 + 2] = i2;
}

// top-3 per column of probs (== rows of probs^T)
__global__ void topk_cols(const float* __restrict__ probs, float* __restrict__ ea,
                          int* __restrict__ ei) {
    long long t = (long long)blockIdx.x * 256 + threadIdx.x;
    if (t >= ROWS) return;
    long long b = t / CS;
    int j = (int)(t % CS);
    const float* base = probs + b * CS * CS + j;
    float v0 = -1.f, v1 = -1.f, v2 = -1.f;
    int   i0 = 0,    i1 = 0,    i2 = 0;
    for (int i = 0; i < CS; ++i) {
        float f = base[(long long)i * CS];
        if (f > v0)      { v2 = v1; i2 = i1; v1 = v0; i1 = i0; v0 = f; i0 = i; }
        else if (f > v1) { v2 = v1; i2 = i1; v1 = f;  i1 = i; }
        else if (f > v2) { v2 = f;  i2 = i; }
    }
    ea[t * 3] = v0; ea[t * 3 + 1] = v1; ea[t * 3 + 2] = v2;
    ei[t * 3] = i0; ei[t * 3 + 1] = i1; ei[t * 3 + 2] = i2;
}

// one wave per (b,e,h): GATv2 edge score + segment-max via encoded atomicMax
__global__ __launch_bounds__(256)
void gat_edge_score(const float* __restrict__ xl, const float* __restrict__ xr,
                    const float* __restrict__ ea, const int* __restrict__ tgt,
                    const float* __restrict__ We, const float* __restrict__ attv,
                    float* __restrict__ sbuf, unsigned* __restrict__ smax,
                    int F, int HF) {
    long long w = (long long)blockIdx.x * 8 + (threadIdx.x >> 5);
    const long long total = (long long)CB * CE * CH;
    if (w >= total) return;
    int lane = threadIdx.x & 31;
    int h = (int)(w % CH);
    long long t = w / CH;
    int e = (int)(t % CE);
    int b = (int)(t / CE);
    int srcn = e / CK;
    int tg   = tgt[(long long)b * CE + e];
    float eav = ea[(long long)b * CE + e];
    const float* xlp = xl + ((long long)b * CS + srcn) * HF + (long long)h * F;
    const float* xrp = xr + ((long long)b * CS + tg)   * HF + (long long)h * F;
    const float* wep = We   + (long long)h * F;
    const float* avp = attv + (long long)h * F;
    float acc = 0.0f;
    for (int c = lane; c < F; c += 32) {
        float m = xlp[c] + xrp[c] + eav * wep[c];
        m = (m > 0.0f) ? m : 0.2f * m;          // leaky_relu 0.2
        acc += m * avp[c];
    }
    acc = wave_sum_f(acc);
    if (lane == 0) {
        sbuf[w] = acc;
        atomicMax(&smax[((long long)b * CS + tg) * CH + h], enc_f(acc));
    }
}

__global__ void gat_edge_soft(const float* __restrict__ sbuf,
                              const unsigned* __restrict__ smax,
                              const int* __restrict__ tgt,
                              float* __restrict__ exbuf, float* __restrict__ den) {
    long long i = (long long)blockIdx.x * 256 + threadIdx.x;
    const long long total = (long long)CB * CE * CH;
    if (i >= total) return;
    int h = (int)(i % CH);
    long long t = i / CH;
    int e = (int)(t % CE);
    int b = (int)(t / CE);
    int tg = tgt[(long long)b * CE + e];
    float ex = expf(sbuf[i] - dec_f(smax[((long long)b * CS + tg) * CH + h]));
    exbuf[i] = ex;
    atomicAdd(&den[((long long)b * CS + tg) * CH + h], ex);
}

__global__ void gat_scatter(const float* __restrict__ xl, const float* __restrict__ exbuf,
                            const float* __restrict__ den, const int* __restrict__ tgt,
                            float* __restrict__ gout, int F, int HF) {
    long long i = (long long)blockIdx.x * 256 + threadIdx.x;
    long long total = (long long)CB * CE * CH * F;
    if (i >= total) return;
    int c = (int)(i % F);
    long long t = i / F;
    int h = (int)(t % CH);
    long long t2 = t / CH;
    int e = (int)(t2 % CE);
    int b = (int)(t2 / CE);
    int tg = tgt[(long long)b * CE + e];
    int srcn = e / CK;
    long long eh = ((long long)b * CE + e) * CH + h;
    float alpha = exbuf[eh] / den[((long long)b * CS + tg) * CH + h];
    atomicAdd(&gout[((long long)b * CS + tg) * F + c],
              alpha * xl[((long long)b * CS + srcn) * HF + (long long)h * F + c]);
}

// xs = tanh(gout/H + res)  (writes f32 master + bf16 mirror)
__global__ void gat_combine(const float* __restrict__ gout, const float* __restrict__ res,
                            float* __restrict__ xs, bfloat16_t* __restrict__ xsb, int F) {
    long long i = (long long)blockIdx.x * 256 + threadIdx.x;
    long long total = ROWS * (long long)F;
    if (i >= total) return;
    float v = tanhf(gout[i] * (1.0f / CH) + res[i]);
    xs[i] = v;
    xsb[i] = (bfloat16_t)v;
}

__global__ void reduce_mask(const float* __restrict__ mf, float* __restrict__ valid) {
    long long i = (long long)blockIdx.x * 256 + threadIdx.x;
    if (i < ROWS) atomicAdd(valid, mf[i]);
}

__global__ void finalize_losses(const float* __restrict__ scal, float* __restrict__ out) {
    if (threadIdx.x == 0 && blockIdx.x == 0) {
        float valid = scal[0] - (float)CB;
        out[0] = scal[1] / valid;
        out[1] = scal[2] / valid;
    }
}

// ---------------------------------------------------------------------------
// host orchestration
// ---------------------------------------------------------------------------
extern "C" void kernel_launch(void* const* d_in, const int* in_sizes, int n_in,
                              void* d_out, int out_size, void* d_ws, size_t ws_size,
                              hipStream_t stream)
{
    (void)in_sizes; (void)out_size; (void)ws_size;
    if (n_in < 78) return;

    auto F32 = [&](int i) { return (const float*)d_in[i]; };
    auto I32 = [&](int i) { return (const int*)d_in[i]; };

    // input mapping: params dict flattened in insertion order
    // 0:x 1:sentinel 2:haf_w 3:haf_b 4:daf_w 5:daf_b 6:htf_w 7:htf_b
    // 8:dtf_w 9:dtf_b 10:bil_W 11:bil_b
    // 12..27 c1a[0..1]{Wl,bl,Wr,br,We,att,Wres,bias} 28..43 c2a  44..59 c1r
    // 60..75 c2r  76:mask 77:head_indices
    struct GatP { const float *Wl, *bl, *Wr, *br, *We, *att, *Wres, *bias; };
    auto loadGat = [&](int b) {
        GatP g;
        g.Wl = F32(b); g.bl = F32(b + 1); g.Wr = F32(b + 2); g.br = F32(b + 3);
        g.We = F32(b + 4); g.att = F32(b + 5); g.Wres = F32(b + 6); g.bias = F32(b + 7);
        return g;
    };
    GatP c1a[2] = { loadGat(12), loadGat(20) };
    GatP c2a[2] = { loadGat(28), loadGat(36) };
    GatP c1r[2] = { loadGat(44), loadGat(52) };
    GatP c2r[2] = { loadGat(60), loadGat(68) };
    const float* xin  = F32(0);
    const float* sent = F32(1);
    const float* bilW = F32(10);
    const float* bilB = F32(11);

    // workspace bump allocator
    size_t off = 0;
    auto allocRaw = [&](long long nbytes) {
        off = (off + 255) & ~(size_t)255;
        void* p = (char*)d_ws + off;
        off += (size_t)nbytes;
        return p;
    };
    auto allocF  = [&](long long n) { return (float*)allocRaw(n * 4); };
    auto allocI  = [&](long long n) { return (int*)allocRaw(n * 4); };
    auto allocU  = [&](long long n) { return (unsigned*)allocRaw(n * 4); };
    auto allocBF = [&](long long n) { return (bfloat16_t*)allocRaw(n * 2); };

    // f32 state
    float* mf   = allocF(ROWS);
    int*   hi   = allocI(ROWS);
    float* scal = allocF(4);                 // [validSum, loss0, loss1, spare]
    float* ha = allocF(ROWS * CA); float* da = allocF(ROWS * CA);
    float* ht = allocF(ROWS * CT); float* dt = allocF(ROWS * CT);
    float* att  = allocF(ROWS * CS);
    float* prob = allocF(ROWS * CS);
    float* ea  = allocF((long long)CB * CE); int* ei  = allocI((long long)CB * CE);
    float* eaT = allocF((long long)CB * CE); int* eiT = allocI((long long)CB * CE);
    float* xl = allocF(ROWS * 2048); float* xr = allocF(ROWS * 2048);
    float* res = allocF(ROWS * CA); float* gout = allocF(ROWS * CA);
    float* sbuf = allocF((long long)CB * CE * CH);
    float* exbf = allocF((long long)CB * CE * CH);
    unsigned* smax = allocU(ROWS * CH);
    float* den = allocF(ROWS * CH);

    // bf16 mirrors (+PADR rows so tile over-reads stay in-bounds)
    bfloat16_t* xab = allocBF((ROWS + PADR) * CD);
    bfloat16_t* hab = allocBF((ROWS + PADR) * CA);
    bfloat16_t* dab = allocBF((ROWS + PADR) * CA);
    bfloat16_t* htb = allocBF((ROWS + PADR) * CT);
    bfloat16_t* dtb = allocBF((ROWS + PADR) * CT);
    bfloat16_t* x1b = allocBF((ROWS + PADR) * CFP);
    bfloat16_t* y1b = allocBF((ROWS + PADR) * CFP);
    bfloat16_t* tpb = allocBF((ROWS + PADR) * CFP);

    auto nb = [](long long n, int b) { return dim3((unsigned)((n + b - 1) / b)); };
    auto convT = [&](const float* src, bfloat16_t* dst, int K, int N, int ldbt) {
        conv_transpose_bf16<<<nb((long long)N * ldbt, 256), 256, 0, stream>>>(
            src, dst, K, N, ldbt);
    };

    // transposed bf16 weight mirrors ([N,K] row-major; N mult of 64 except bilW
    // which gets 576-row slices; K-pad zero-filled by conv_transpose)
    bfloat16_t* hafT = allocBF((long long)CA * CD);
    bfloat16_t* dafT = allocBF((long long)CA * CD);
    bfloat16_t* htfT = allocBF((long long)CT * CD);
    bfloat16_t* dtfT = allocBF((long long)CT * CD);
    bfloat16_t* bilWT = allocBF(3LL * LBS);
    struct GatW { bfloat16_t *WlT, *WrT, *WresT; };
    GatW wArc1[2], wArc2[2], wTag1[2], wTag2[2];
    auto makeGatW = [&](const GatP& p, int F) {
        GatW w; int HF = 4 * F;
        w.WlT   = allocBF((long long)HF * F);
        w.WrT   = allocBF((long long)HF * F);
        w.WresT = allocBF((long long)F * F);
        convT(p.Wl,   w.WlT,   F, HF, F);
        convT(p.Wr,   w.WrT,   F, HF, F);
        convT(p.Wres, w.WresT, F, F,  F);
        return w;
    };

    convT(F32(2), hafT, CD, CA, CD);
    convT(F32(4), dafT, CD, CA, CD);
    convT(F32(6), htfT, CD, CT, CD);
    convT(F32(8), dtfT, CD, CT, CD);
    for (int l = 0; l < 3; ++l)
        convT(bilW + (long long)l * 513 * 513, bilWT + (long long)l * LBS,
              513, 513, CFP);
    for (int l = 0; l < 2; ++l) {
        wArc1[l] = makeGatW(c1a[l], CA);
        wArc2[l] = makeGatW(c2a[l], CA);
        wTag1[l] = makeGatW(c1r[l], CT);
        wTag2[l] = makeGatW(c2r[l], CT);
    }
    // tpb K-pad columns (513..543) are never written by GEMM#1 -> zero them
    fill_bf16<<<nb((ROWS + PADR) * CFP, 256), 256, 0, stream>>>(
        tpb, 0.0f, (ROWS + PADR) * CFP);

    auto gemm = [&](const bfloat16_t* Ap, const bfloat16_t* Bp, const float* biasp,
                    float* Cp, bfloat16_t* Cbp,
                    int M, int N, int Kd, int lda, int ldb, int ldc,
                    long long sA, long long sB, long long sC, int batches,
                    int biasMode, int act) {
        dim3 grid((unsigned)((N + TN - 1) / TN), (unsigned)((M + TM - 1) / TM),
                  (unsigned)batches);
        gemm_wmma_bf16<<<grid, dim3(256), 0, stream>>>(
            Ap, Bp, biasp, Cp, Cbp, M, N, Kd, lda, ldb, ldc, sA, sB, sC,
            biasMode, act);
    };

    const int Mr = (int)ROWS;
    const long long sPad = (long long)CS * CFP;
    const long long sAtt = (long long)CS * CS;

    // stage 0: augment inputs + valid count
    build_xa<<<nb(ROWS * CD, 256), 256, 0, stream>>>(xin, sent, xab);
    build_aux<<<nb(ROWS, 256), 256, 0, stream>>>(I32(76), I32(77), mf, hi);
    fill_f32<<<1, 256, 0, stream>>>(scal, 0.0f, 4);
    reduce_mask<<<nb(ROWS, 256), 256, 0, stream>>>(mf, scal);

    // stage 1: projections, fused ELU, dual f32+bf16 outputs
    gemm(xab, hafT, F32(3), ha, hab, Mr, CA, CD, CD, CD, CA, 0, 0, 0, 1, 1, 1);
    gemm(xab, dafT, F32(5), da, dab, Mr, CA, CD, CD, CD, CA, 0, 0, 0, 1, 1, 1);
    gemm(xab, htfT, F32(7), ht, htb, Mr, CT, CD, CD, CD, CT, 0, 0, 0, 1, 1, 1);
    gemm(xab, dtfT, F32(9), dt, dtb, Mr, CT, CD, CD, CD, CT, 0, 0, 0, 1, 1, 1);

    for (int l = 0; l < 2; ++l) {
        // bilinear att = [ha,1] @ W_l @ [da,1]^T (batched per sample)
        pad_ones<<<nb(ROWS * CFP, 256), 256, 0, stream>>>(ha, x1b, CA, CFP, ROWS);
        pad_ones<<<nb(ROWS * CFP, 256), 256, 0, stream>>>(da, y1b, CA, CFP, ROWS);
        gemm(x1b, bilWT + (long long)l * LBS, nullptr, nullptr, tpb,
             CS, 513, 513, CFP, CFP, CFP, sPad, 0, sPad, CB, 0, 0);
        gemm(tpb, y1b, bilB + l, att, nullptr,
             CS, CS, 513, CFP, CFP, CS, sPad, sPad, sAtt, CB, 2, 0);

        loss_rows<<<nb(ROWS, 8), 256, 0, stream>>>(att, mf, hi, scal + 1 + l);
        row_softmax<<<nb(ROWS, 8), 256, 0, stream>>>(att, prob);
        topk_rows<<<nb(ROWS, 256), 256, 0, stream>>>(prob, ea, ei);
        topk_cols<<<nb(ROWS, 256), 256, 0, stream>>>(prob, eaT, eiT);

        // 4 GATv2 streams
        struct SD { float* xs; bfloat16_t* xsb; const GatP* P; const GatW* W;
                    const float* eaP; const int* tgtP; int F; };
        SD sd[4] = { { ha, hab, &c1a[l], &wArc1[l], ea,  ei,  CA },
                     { da, dab, &c2a[l], &wArc2[l], eaT, eiT, CA },
                     { ht, htb, &c1r[l], &wTag1[l], ea,  ei,  CT },
                     { dt, dtb, &c2r[l], &wTag2[l], eaT, eiT, CT } };
        for (int s = 0; s < 4; ++s) {
            const int F = sd[s].F, HF = 4 * F;
            const GatP* P = sd[s].P;
            const GatW* W = sd[s].W;
            gemm(sd[s].xsb, W->WlT,   P->bl,   xl,  nullptr, Mr, HF, F, F, F, HF, 0, 0, 0, 1, 1, 0);
            gemm(sd[s].xsb, W->WrT,   P->br,   xr,  nullptr, Mr, HF, F, F, F, HF, 0, 0, 0, 1, 1, 0);
            gemm(sd[s].xsb, W->WresT, P->bias, res, nullptr, Mr, F,  F, F, F, F,  0, 0, 0, 1, 1, 0);
            fill_u32<<<nb(ROWS * CH, 256), 256, 0, stream>>>(smax, 0u, ROWS * CH);
            fill_f32<<<nb(ROWS * CH, 256), 256, 0, stream>>>(den, 0.0f, ROWS * CH);
            fill_f32<<<nb(ROWS * F, 256), 256, 0, stream>>>(gout, 0.0f, ROWS * F);
            gat_edge_score<<<nb((long long)CB * CE * CH, 8), 256, 0, stream>>>(
                xl, xr, sd[s].eaP, sd[s].tgtP, P->We, P->att, sbuf, smax, F, HF);
            gat_edge_soft<<<nb((long long)CB * CE * CH, 256), 256, 0, stream>>>(
                sbuf, smax, sd[s].tgtP, exbf, den);
            gat_scatter<<<nb((long long)CB * CE * CH * F, 256), 256, 0, stream>>>(
                xl, exbf, den, sd[s].tgtP, gout, F, HF);
            gat_combine<<<nb(ROWS * F, 256), 256, 0, stream>>>(
                gout, res, sd[s].xs, sd[s].xsb, F);
        }
    }

    // final bilinear -> attended directly into d_out
    float* dout = (float*)d_out;
    const long long ATT_SZ = (long long)CB * CS * CS;
    const long long TAG_SZ = (long long)CB * CS * CT;
    pad_ones<<<nb(ROWS * CFP, 256), 256, 0, stream>>>(ha, x1b, CA, CFP, ROWS);
    pad_ones<<<nb(ROWS * CFP, 256), 256, 0, stream>>>(da, y1b, CA, CFP, ROWS);
    gemm(x1b, bilWT + 2LL * LBS, nullptr, nullptr, tpb,
         CS, 513, 513, CFP, CFP, CFP, sPad, 0, sPad, CB, 0, 0);
    gemm(tpb, y1b, bilB + 2, dout, nullptr,
         CS, CS, 513, CFP, CFP, CS, sPad, sPad, sAtt, CB, 2, 0);

    copy_f32<<<nb(TAG_SZ, 256), 256, 0, stream>>>(dout + ATT_SZ, ht, TAG_SZ);
    copy_f32<<<nb(TAG_SZ, 256), 256, 0, stream>>>(dout + ATT_SZ + TAG_SZ, dt, TAG_SZ);
    finalize_losses<<<1, 32, 0, stream>>>(scal, dout + ATT_SZ + 2 * TAG_SZ);
}
